// DRNN_24902220382314
// MI455X (gfx1250) — compile-verified, matrix operand order checked by reading, not run on hardware
//
#include <hip/hip_runtime.h>

// Problem constants (from the reference)
constexpr int kT = 8192;   // timesteps (GEMM M)
constexpr int kK = 1024;   // input dim  (GEMM K)
constexpr int kR = 4096;   // reservoir  (GEMM N)

typedef __attribute__((ext_vector_type(16))) __bf16 v16bf;
typedef __attribute__((ext_vector_type(8)))  __bf16 v8bf;
typedef __attribute__((ext_vector_type(4)))  __bf16 v4bf;
typedef __attribute__((ext_vector_type(8)))  float  v8f;

// Tiling
constexpr int BM = 64;          // rows of X per block
constexpr int BN = 64;          // reservoir cols per block
constexpr int BK = 32;          // K step (16x16x32 bf16 WMMA)
constexpr int LSTR = 40;        // bf16 row stride: 80B = 20 dwords -> conflict-free b128 gathers

// ---------------------------------------------------------------------------
// Kernel 1: U = X @ W^T via split-bf16 WMMA (hi/lo, 3 MMAs ~ f32 accuracy).
// f32 -> bf16(hi,lo) conversion happens ONCE at LDS staging; the inner loop is
// pure ds_load_b128 + v_wmma.
// ---------------------------------------------------------------------------
__global__ __launch_bounds__(128) void drnn_gemm(const float* __restrict__ X,
                                                 const float* __restrict__ W,
                                                 float* __restrict__ U) {
    __shared__ __bf16 AsHi[BM][LSTR];
    __shared__ __bf16 AsLo[BM][LSTR];
    __shared__ __bf16 BsHi[BN][LSTR];
    __shared__ __bf16 BsLo[BN][LSTR];

    const int tid  = threadIdx.x;
    const int lane = tid & 31;
    const int wave = tid >> 5;              // 0..3, each owns a 16-row M slab
    const int m0   = blockIdx.y * BM;
    const int n0   = blockIdx.x * BN;

    const int rloc  = lane & 15;            // row within 16 (A: M, B: N)
    const int khalf = (lane >> 4) << 3;     // lanes 16-31 handle K+8 / K+24

    v8f acc[4];
    #pragma unroll
    for (int s = 0; s < 4; ++s) {
        #pragma unroll
        for (int v = 0; v < 8; ++v) acc[s][v] = 0.0f;
    }

    for (int k0 = 0; k0 < kK; k0 += BK) {
        // ---- stage + split-convert: global f32 -> LDS bf16 hi/lo ----
        #pragma unroll
        for (int i = tid; i < BM * 8; i += 128) {
            const int r = i >> 3;
            const int c = (i & 7) << 2;
            const float4 va = *(const float4*)(X + (size_t)(m0 + r) * kK + k0 + c);
            const float4 vb = *(const float4*)(W + (size_t)(n0 + r) * kK + k0 + c);

            v4bf ah = {(__bf16)va.x, (__bf16)va.y, (__bf16)va.z, (__bf16)va.w};
            v4bf al = {(__bf16)(va.x - (float)ah[0]), (__bf16)(va.y - (float)ah[1]),
                       (__bf16)(va.z - (float)ah[2]), (__bf16)(va.w - (float)ah[3])};
            v4bf bh = {(__bf16)vb.x, (__bf16)vb.y, (__bf16)vb.z, (__bf16)vb.w};
            v4bf bl = {(__bf16)(vb.x - (float)bh[0]), (__bf16)(vb.y - (float)bh[1]),
                       (__bf16)(vb.z - (float)bh[2]), (__bf16)(vb.w - (float)bh[3])};

            *(v4bf*)&AsHi[r][c] = ah;
            *(v4bf*)&AsLo[r][c] = al;
            *(v4bf*)&BsHi[r][c] = bh;
            *(v4bf*)&BsLo[r][c] = bl;
        }
        __syncthreads();

        // ---- A fragments: each half-fragment is one ds_load_b128 ----
        const int arow = wave * 16 + rloc;
        const v8bf ah0 = *(const v8bf*)&AsHi[arow][khalf];
        const v8bf ah1 = *(const v8bf*)&AsHi[arow][khalf + 16];
        const v8bf al0 = *(const v8bf*)&AsLo[arow][khalf];
        const v8bf al1 = *(const v8bf*)&AsLo[arow][khalf + 16];
        const v16bf ahi = __builtin_shufflevector(ah0, ah1, 0, 1, 2, 3, 4, 5, 6, 7,
                                                  8, 9, 10, 11, 12, 13, 14, 15);
        const v16bf alo = __builtin_shufflevector(al0, al1, 0, 1, 2, 3, 4, 5, 6, 7,
                                                  8, 9, 10, 11, 12, 13, 14, 15);

        // ---- 4 N-subtiles: 4 b128 loads + 3 WMMAs each ----
        #pragma unroll
        for (int s = 0; s < 4; ++s) {
            const int brow = s * 16 + rloc;
            const v8bf bh0 = *(const v8bf*)&BsHi[brow][khalf];
            const v8bf bh1 = *(const v8bf*)&BsHi[brow][khalf + 16];
            const v8bf bl0 = *(const v8bf*)&BsLo[brow][khalf];
            const v8bf bl1 = *(const v8bf*)&BsLo[brow][khalf + 16];
            const v16bf bhi = __builtin_shufflevector(bh0, bh1, 0, 1, 2, 3, 4, 5, 6, 7,
                                                      8, 9, 10, 11, 12, 13, 14, 15);
            const v16bf blo = __builtin_shufflevector(bl0, bl1, 0, 1, 2, 3, 4, 5, 6, 7,
                                                      8, 9, 10, 11, 12, 13, 14, 15);

            acc[s] = __builtin_amdgcn_wmma_f32_16x16x32_bf16(
                false, ahi, false, bhi, (short)0, acc[s], false, false);
            acc[s] = __builtin_amdgcn_wmma_f32_16x16x32_bf16(
                false, ahi, false, blo, (short)0, acc[s], false, false);
            acc[s] = __builtin_amdgcn_wmma_f32_16x16x32_bf16(
                false, alo, false, bhi, (short)0, acc[s], false, false);
        }
        __syncthreads();
    }

    // ---- store C/D (lanes 0-15: M=v; lanes 16-31: M=v+8); keep RT so U stays in L2 ----
    #pragma unroll
    for (int s = 0; s < 4; ++s) {
        const int n = n0 + s * 16 + rloc;
        #pragma unroll
        for (int v = 0; v < 8; ++v) {
            const int m = m0 + wave * 16 + v + ((lane >> 4) << 3);
            U[(size_t)m * kR + n] = acc[s][v];
        }
    }
}

// ---------------------------------------------------------------------------
// Kernel 2: diagonal-reservoir scan. Latency-bound: 8192 dependent steps.
// Chain per step = v_fmac_f32 + v_tanh_f32 (2 ops); 16-deep register
// prefetch ring keeps ~16 loads in flight to hide L2 latency (U is
// L2-resident right after the GEMM: 134 MB < 192 MB L2).
// ---------------------------------------------------------------------------
__device__ __forceinline__ float my_tanhf(float x) {
#if __has_builtin(__builtin_amdgcn_tanhf)
    return __builtin_amdgcn_tanhf(x);            // v_tanh_f32 (CDNA5 trans op)
#else
    const float e = __builtin_amdgcn_exp2f(x * 2.8853900817779268f);
    return 1.0f - 2.0f * __builtin_amdgcn_rcpf(e + 1.0f);
#endif
}

constexpr int PF = 16;  // prefetch depth (register ring)

__global__ __launch_bounds__(128) void drnn_scan(float* __restrict__ U,
                                                 const float* __restrict__ d) {
    const int r = blockIdx.x * blockDim.x + threadIdx.x;   // 0..kR-1
    const float dr = d[r];

    float buf[PF];
    #pragma unroll
    for (int i = 0; i < PF; ++i) buf[i] = U[(size_t)i * kR + r];

    float s = 0.0f;
    for (int t0 = 0; t0 < kT; t0 += PF) {
        #pragma unroll
        for (int j = 0; j < PF; ++j) {
            const int t = t0 + j;
            const float u = buf[j];
            if (t + PF < kT) buf[j] = U[(size_t)(t + PF) * kR + r];  // uniform branch
            s = my_tanhf(fmaf(dr, s, u));
            U[(size_t)t * kR + r] = s;
        }
    }
}

// ---------------------------------------------------------------------------
extern "C" void kernel_launch(void* const* d_in, const int* in_sizes, int n_in,
                              void* d_out, int out_size, void* d_ws, size_t ws_size,
                              hipStream_t stream) {
    (void)in_sizes; (void)n_in; (void)out_size; (void)d_ws; (void)ws_size;
    const float* X = (const float*)d_in[0];   // [T, K, 1]
    const float* W = (const float*)d_in[1];   // [R, K]
    const float* d = (const float*)d_in[2];   // [R]
    float* out = (float*)d_out;               // [T, R, 1]

    dim3 grid(kR / BN, kT / BM);              // 64 x 128 workgroups
    drnn_gemm<<<grid, 128, 0, stream>>>(X, W, out);
    drnn_scan<<<kR / 128, 128, 0, stream>>>(out, d);
}
